// SequencePredictionLayer_42099269435819
// MI455X (gfx1250) — matrix operand order repaired and checked by reading, not compiled, not run on hardware
//
#include <hip/hip_runtime.h>
#include <hip/hip_bf16.h>

// Problem constants (from reference): B=48, L=192, I=H=1024, V=8192, E=10000
#define BB   48
#define LL   192
#define TT   193          // L+1 timesteps
#define II   1024
#define HH   1024
#define VV   8192
#define H3   3072         // 3*H
#define MM   (TT * BB)    // 9264 flattened (t,b) rows

#define LOG_EPS (-102.9415f)   // logf(1e-45f) after f32 rounding (1.4013e-45)

typedef __attribute__((ext_vector_type(16))) __bf16 bf16x16;
typedef __attribute__((ext_vector_type(8)))  __bf16 bf16x8;
typedef __attribute__((ext_vector_type(8)))  float  f32x8;
typedef int vi4 __attribute__((vector_size(16)));   // b128 payload type

#define AS1 __attribute__((address_space(1)))
#define AS3 __attribute__((address_space(3)))

// CDNA5 async global->LDS copy (ASYNCcnt path); fall back to load+ds_store.
#if defined(__HIP_DEVICE_COMPILE__) && \
    __has_builtin(__builtin_amdgcn_global_load_async_to_lds_b128)
#define HAVE_ASYNC_LDS 1
#else
#define HAVE_ASYNC_LDS 0
#endif

__device__ __forceinline__ void copy16_g2l(const void* g, void* l) {
#if HAVE_ASYNC_LDS
  __builtin_amdgcn_global_load_async_to_lds_b128(
      (AS1 vi4*)g, (AS3 vi4*)l, 0, 0);
#else
  *(bf16x8*)l = *(const bf16x8*)g;
#endif
}

__device__ __forceinline__ void wait_async0() {
#if defined(__HIP_DEVICE_COMPILE__) && \
    __has_builtin(__builtin_amdgcn_s_wait_asynccnt)
  __builtin_amdgcn_s_wait_asynccnt(0);
#elif HAVE_ASYNC_LDS
  asm volatile("s_wait_asynccnt 0" ::: "memory");
#endif
}

__device__ __forceinline__ f32x8 zero8() {
  f32x8 v;
#pragma unroll
  for (int i = 0; i < 8; ++i) v[i] = 0.f;
  return v;
}

__device__ __forceinline__ bf16x16 cat8(bf16x8 lo, bf16x8 hi) {
  return __builtin_shufflevector(lo, hi, 0,1,2,3,4,5,6,7,8,9,10,11,12,13,14,15);
}

__device__ __forceinline__ f32x8 wmma_bf16(bf16x16 a, bf16x16 b, f32x8 c) {
  // D = A(16x32) * B(32x16) + C, f32 accumulate. 8-arg CDNA5 form.
  return __builtin_amdgcn_wmma_f32_16x16x32_bf16(false, a, false, b,
                                                 (short)0, c, false, false);
}

__device__ __forceinline__ float sigmoidf_(float x) {
  return 1.f / (1.f + expf(-x));
}

// ---------------------------------------------------------------- elementwise
__global__ void f32_to_bf16_k(const float* __restrict__ src,
                              __bf16* __restrict__ dst, int n) {
  int i = blockIdx.x * blockDim.x + threadIdx.x;
  if (i < n) dst[i] = (__bf16)src[i];
}

// xs[t,b,:] = (t==0) ? start : user_table[seq[b,t-1]]   (stored bf16)
__global__ void build_xs_k(const float* __restrict__ start,
                           const float* __restrict__ user_table,
                           const int* __restrict__ seq,
                           __bf16* __restrict__ xs) {
  int idx = blockIdx.x * blockDim.x + threadIdx.x;
  if (idx >= MM * II) return;
  int t = idx / (BB * II);
  int r = idx - t * (BB * II);
  int b = r >> 10;
  int i = r & (II - 1);
  float v = (t == 0) ? start[i]
                     : user_table[(size_t)seq[b * LL + (t - 1)] * II + i];
  xs[idx] = (__bf16)v;
}

// h0 = event_table[ids]  -> f32 state + bf16 copy for WMMA
__global__ void gather_h0_k(const float* __restrict__ event_table,
                            const int* __restrict__ ids,
                            float* __restrict__ hf, __bf16* __restrict__ hb) {
  int i = blockIdx.x * blockDim.x + threadIdx.x;
  if (i >= BB * HH) return;
  int b = i >> 10;
  float v = event_table[(size_t)ids[b] * HH + (i & (HH - 1))];
  hf[i] = v;
  hb[i] = (__bf16)v;
}

// ---------------------------------------------------------------- batched GEMM
// C[M,N] = A[M,K](bf16,row-major) * W[N,K](bf16,row-major)^T + bias[N]  (f32 out)
// Block tile 64(M) x 128(N); 8 waves; wave = 16x64 (4 WMMA tiles); K step 32.
// A/B tiles staged via async global->LDS DMA (ASYNCcnt), frags read with
// ds_load_b128. permuteTB: out row for m=(t*BB+b) becomes b*TT+t.
__global__ __launch_bounds__(256)
void gemm_bf16_wmma(const __bf16* __restrict__ A, int lda, int M,
                    const __bf16* __restrict__ W, int ldw,
                    const float* __restrict__ bias,
                    float* __restrict__ C, int ldc, int K, int permuteTB) {
  __shared__ __align__(16) __bf16 sA[64 * 32];
  __shared__ __align__(16) __bf16 sB[128 * 32];

  const int tid   = threadIdx.x;
  const int lane  = tid & 31;
  const int wave  = tid >> 5;
  const int mtile = wave & 3;     // 4 M-tiles of 16
  const int ngrp  = wave >> 2;    // 2 N-groups of 64
  const int m0    = blockIdx.x * 64 + mtile * 16;
  const int nbase = blockIdx.y * 128 + ngrp * 64;
  const int row16 = lane & 15;
  const int kh    = lane >> 4;

  f32x8 acc[4];
#pragma unroll
  for (int i = 0; i < 4; ++i) acc[i] = zero8();

  // staging coordinates (per thread)
  const int ar = tid >> 2, ac = (tid & 3) * 8;        // A: 256 chunks of 16B
  const int br = tid >> 1, bc = (tid & 1) * 16;       // B: 2x 16B per thread
  int agr = blockIdx.x * 64 + ar;
  if (agr >= M) agr = M - 1;                          // clamp tail reads
  const size_t abase = (size_t)agr * lda + ac;
  const size_t bbase = (size_t)(blockIdx.y * 128 + br) * ldw + bc;

  for (int k0 = 0; k0 < K; k0 += 32) {
    copy16_g2l(&A[abase + k0],     &sA[ar * 32 + ac]);
    copy16_g2l(&W[bbase + k0],     &sB[br * 32 + bc]);
    copy16_g2l(&W[bbase + k0 + 8], &sB[br * 32 + bc + 8]);
    if (k0 + 32 < K) {                                // pull next K tile near
      __builtin_prefetch(&A[abase + k0 + 32], 0, 3);
      __builtin_prefetch(&W[bbase + k0 + 32], 0, 3);
    }
    wait_async0();
    __syncthreads();

    // A fragment: lane(row16,kh) -> K {kh*8..+7} and {16+kh*8..+7}
    const __bf16* ap = &sA[(mtile * 16 + row16) * 32];
    bf16x16 af = cat8(*(const bf16x8*)(ap + kh * 8),
                      *(const bf16x8*)(ap + 16 + kh * 8));
#pragma unroll
    for (int nt = 0; nt < 4; ++nt) {
      // B fragment: lane holds column (lane&15), K {kh*8..+7} and {16+kh*8..+7}
      const __bf16* bp = &sB[(ngrp * 64 + nt * 16 + row16) * 32 + kh * 8];
      bf16x16 bf_ = cat8(*(const bf16x8*)bp, *(const bf16x8*)(bp + 16));
      acc[nt] = wmma_bf16(af, bf_, acc[nt]);
    }
    __syncthreads();
  }

#pragma unroll
  for (int nt = 0; nt < 4; ++nt) {
    const int col = nbase + nt * 16 + row16;
    const float bv = bias[col];
#pragma unroll
    for (int r = 0; r < 8; ++r) {
      const int m = m0 + r + kh * 8;
      if (m < M) {
        size_t orow = permuteTB ? ((size_t)(m % BB) * TT + (m / BB)) : (size_t)m;
        C[orow * (size_t)ldc + col] = acc[nt][r] + bv;
      }
    }
  }
}

// ---------------------------------------------------------------- GRU step
// One block per 16 hidden columns (j0). Each block computes, for all 48 batch
// rows, the r/z/n slices of hp = h @ w_hh^T, reduces K across 4 waves via LDS
// f32 atomics, then applies the fused GRU gate math.
__global__ __launch_bounds__(128)
void gru_step_k(const __bf16* __restrict__ hbf,   // [48,1024] old h (bf16)
                const float*  __restrict__ hf,    // [48,1024] old h (f32)
                const __bf16* __restrict__ whh,   // [3072,1024] bf16
                const float*  __restrict__ bhh,   // [3072]
                const float*  __restrict__ xp,    // [TT,48,3072] (has b_ih)
                int t,
                float*  __restrict__ hf_new,
                __bf16* __restrict__ hbf_new,
                __bf16* __restrict__ ys) {        // [TT,48,1024]
  __shared__ float red[9 * 256];                  // [gate*3+mtile][16x16]
  const int tid  = threadIdx.x;
  const int lane = tid & 31;
  const int wave = tid >> 5;
  const int row16 = lane & 15;
  const int kh    = lane >> 4;
  const int j0 = blockIdx.x * 16;

  for (int i = tid; i < 9 * 256; i += 128) red[i] = 0.f;
  __syncthreads();

  f32x8 acc[9];
#pragma unroll
  for (int i = 0; i < 9; ++i) acc[i] = zero8();

  const int kbase = wave * 256;                   // K split across 4 waves
#pragma unroll 1
  for (int ks = 0; ks < 256; ks += 32) {
    const int k0 = kbase + ks;
    bf16x16 af[3];
#pragma unroll
    for (int mt = 0; mt < 3; ++mt) {
      const __bf16* ap = hbf + (size_t)(mt * 16 + row16) * HH + k0;
      af[mt] = cat8(*(const bf16x8*)(ap + kh * 8),
                    *(const bf16x8*)(ap + 16 + kh * 8));
    }
#pragma unroll
    for (int g = 0; g < 3; ++g) {
      const __bf16* bp = whh + (size_t)(g * HH + j0 + row16) * HH + k0 + kh * 8;
      bf16x16 bf_ = cat8(*(const bf16x8*)bp, *(const bf16x8*)(bp + 16));
#pragma unroll
      for (int mt = 0; mt < 3; ++mt)
        acc[g * 3 + mt] = wmma_bf16(af[mt], bf_, acc[g * 3 + mt]);
    }
  }

  // cross-wave K reduction: C elem r -> (row r+kh*8, col row16)
#pragma unroll
  for (int i9 = 0; i9 < 9; ++i9) {
#pragma unroll
    for (int r = 0; r < 8; ++r) {
      atomicAdd(&red[i9 * 256 + (r + kh * 8) * 16 + row16], acc[i9][r]);
    }
  }
  __syncthreads();

  const float* xpt = xp + (size_t)t * BB * H3;
  for (int i = tid; i < BB * 16; i += 128) {      // 768 outputs
    const int m = i >> 4, c = i & 15;
    const int g = j0 + c;
    const int mt = m >> 4, mr = m & 15;
    const float hr = red[(0 * 3 + mt) * 256 + mr * 16 + c] + bhh[g];
    const float hz = red[(1 * 3 + mt) * 256 + mr * 16 + c] + bhh[HH + g];
    const float hn = red[(2 * 3 + mt) * 256 + mr * 16 + c] + bhh[2 * HH + g];
    const float xr = xpt[m * H3 + g];
    const float xz = xpt[m * H3 + HH + g];
    const float xn = xpt[m * H3 + 2 * HH + g];
    const float rg = sigmoidf_(xr + hr);
    const float zg = sigmoidf_(xz + hz);
    const float ng = tanhf(xn + rg * hn);
    const float hnew = (1.f - zg) * ng + zg * hf[m * HH + g];
    hf_new[m * HH + g] = hnew;
    const __bf16 hb2 = (__bf16)hnew;
    hbf_new[m * HH + g] = hb2;
    ys[(size_t)t * BB * HH + m * HH + g] = hb2;
  }
}

// ---------------------------------------------------------------- mask helpers
__global__ void lastocc_init_k(int* __restrict__ lo) {
  int i = blockIdx.x * blockDim.x + threadIdx.x;
  if (i < BB * VV) lo[i] = -1;
}

__global__ void lastocc_fill_k(const int* __restrict__ seq, int* __restrict__ lo) {
  int i = blockIdx.x * blockDim.x + threadIdx.x;
  if (i >= BB * LL) return;
  int b = i / LL, pos = i - b * LL;
  atomicMax(&lo[b * VV + seq[i]], pos);           // global_atomic_max_i32
}

// ---------------------------------------------------------------- softmax
// One block per (b,t) row of 8192; suffix mask == (lastocc[b][v] >= t) for t<L-1.
__global__ __launch_bounds__(256)
void masked_softmax_k(float* __restrict__ out, const int* __restrict__ lo) {
  __shared__ float s[256];
  const int bt = blockIdx.x;
  const int b = bt / TT, t = bt - b * TT;
  float* row = out + (size_t)bt * VV;
  const int* lob = lo + b * VV;
  const bool maskon = (t < LL - 1);
  const int tid = threadIdx.x;

  float x[VV / 256];
  float mx = -3.4e38f;
#pragma unroll
  for (int j = 0; j < VV / 256; ++j) {
    const int v = tid + j * 256;
    float val = row[v];
    if (maskon && lob[v] < t) val += LOG_EPS;
    x[j] = val;
    mx = fmaxf(mx, val);
  }
  s[tid] = mx; __syncthreads();
  for (int o = 128; o > 0; o >>= 1) {
    if (tid < o) s[tid] = fmaxf(s[tid], s[tid + o]);
    __syncthreads();
  }
  mx = s[0]; __syncthreads();

  float sum = 0.f;
#pragma unroll
  for (int j = 0; j < VV / 256; ++j) {
    x[j] = expf(x[j] - mx);
    sum += x[j];
  }
  s[tid] = sum; __syncthreads();
  for (int o = 128; o > 0; o >>= 1) {
    if (tid < o) s[tid] += s[tid + o];
    __syncthreads();
  }
  const float inv = 1.f / s[0];
#pragma unroll
  for (int j = 0; j < VV / 256; ++j) row[tid + j * 256] = x[j] * inv;
}

// ---------------------------------------------------------------- host driver
extern "C" void kernel_launch(void* const* d_in, const int* in_sizes, int n_in,
                              void* d_out, int out_size, void* d_ws, size_t ws_size,
                              hipStream_t stream) {
  (void)in_sizes; (void)n_in; (void)out_size; (void)ws_size;
  const float* event_table = (const float*)d_in[0];
  const float* user_table  = (const float*)d_in[1];
  const float* start_v     = (const float*)d_in[2];
  const float* w_ih0 = (const float*)d_in[3];
  const float* w_hh0 = (const float*)d_in[4];
  const float* b_ih0 = (const float*)d_in[5];
  const float* b_hh0 = (const float*)d_in[6];
  const float* w_ih1 = (const float*)d_in[7];
  const float* w_hh1 = (const float*)d_in[8];
  const float* b_ih1 = (const float*)d_in[9];
  const float* b_hh1 = (const float*)d_in[10];
  const float* lin_w = (const float*)d_in[11];
  const float* lin_b = (const float*)d_in[12];
  const int* ids = (const int*)d_in[13];
  const int* seq = (const int*)d_in[14];
  float* out = (float*)d_out;

  char* ws = (char*)d_ws;
  size_t off = 0;
  auto alloc = [&](size_t bytes) -> void* {
    void* p = ws + off;
    off += (bytes + 255) & ~(size_t)255;
    return p;
  };
  __bf16* wih0b = (__bf16*)alloc((size_t)H3 * II * 2);
  __bf16* whh0b = (__bf16*)alloc((size_t)H3 * HH * 2);
  __bf16* wih1b = (__bf16*)alloc((size_t)H3 * HH * 2);
  __bf16* whh1b = (__bf16*)alloc((size_t)H3 * HH * 2);
  __bf16* linwb = (__bf16*)alloc((size_t)VV * HH * 2);
  __bf16* xsb   = (__bf16*)alloc((size_t)MM * II * 2);
  float*  xp    = (float*) alloc((size_t)MM * H3 * 4);
  __bf16* ys0b  = (__bf16*)alloc((size_t)MM * HH * 2);
  __bf16* ys1b  = (__bf16*)alloc((size_t)MM * HH * 2);
  float*  hf[2]  = {(float*) alloc((size_t)BB * HH * 4),
                    (float*) alloc((size_t)BB * HH * 4)};
  __bf16* hb[2]  = {(__bf16*)alloc((size_t)BB * HH * 2),
                    (__bf16*)alloc((size_t)BB * HH * 2)};
  int* lastocc = (int*)alloc((size_t)BB * VV * 4);

  // 1) weights -> bf16 (fits entirely in the 192MB L2 for the recurrent loop)
  f32_to_bf16_k<<<(H3 * II + 255) / 256, 256, 0, stream>>>(w_ih0, wih0b, H3 * II);
  f32_to_bf16_k<<<(H3 * HH + 255) / 256, 256, 0, stream>>>(w_hh0, whh0b, H3 * HH);
  f32_to_bf16_k<<<(H3 * HH + 255) / 256, 256, 0, stream>>>(w_ih1, wih1b, H3 * HH);
  f32_to_bf16_k<<<(H3 * HH + 255) / 256, 256, 0, stream>>>(w_hh1, whh1b, H3 * HH);
  f32_to_bf16_k<<<(VV * HH + 255) / 256, 256, 0, stream>>>(lin_w, linwb, VV * HH);

  // 2) xs = [start; user_table[sequence]]  (bf16)
  build_xs_k<<<(MM * II + 255) / 256, 256, 0, stream>>>(start_v, user_table, seq, xsb);

  // 3) xp0 = xs @ w_ih0^T + b_ih0   [9264, 3072]
  dim3 gXP((MM + 63) / 64, H3 / 128);
  gemm_bf16_wmma<<<gXP, 256, 0, stream>>>(xsb, II, MM, wih0b, II, b_ih0,
                                          xp, H3, II, 0);

  // 4) layer-0 recurrence (193 sequential steps, ping-pong h)
  gather_h0_k<<<(BB * HH + 255) / 256, 256, 0, stream>>>(event_table, ids, hf[0], hb[0]);
  int p = 0;
  for (int t = 0; t < TT; ++t) {
    gru_step_k<<<HH / 16, 128, 0, stream>>>(hb[p], hf[p], whh0b, b_hh0, xp, t,
                                            hf[1 - p], hb[1 - p], ys0b);
    p ^= 1;
  }

  // 5) xp1 = ys0 @ w_ih1^T + b_ih1  (reuse xp buffer)
  gemm_bf16_wmma<<<gXP, 256, 0, stream>>>(ys0b, HH, MM, wih1b, HH, b_ih1,
                                          xp, H3, HH, 0);

  // 6) layer-1 recurrence (same h0)
  gather_h0_k<<<(BB * HH + 255) / 256, 256, 0, stream>>>(event_table, ids, hf[0], hb[0]);
  p = 0;
  for (int t = 0; t < TT; ++t) {
    gru_step_k<<<HH / 16, 128, 0, stream>>>(hb[p], hf[p], whh1b, b_hh1, xp, t,
                                            hf[1 - p], hb[1 - p], ys1b);
    p ^= 1;
  }

  // 7) logits = ys1 @ lin_w^T + lin_b, written permuted [b, t, v] into d_out
  dim3 gLG((MM + 63) / 64, VV / 128);
  gemm_bf16_wmma<<<gLG, 256, 0, stream>>>(ys1b, HH, MM, linwb, HH, lin_b,
                                          out, VV, HH, 1);

  // 8) suffix mask via last-occurrence, then fused masked softmax in-place
  lastocc_init_k<<<(BB * VV + 255) / 256, 256, 0, stream>>>(lastocc);
  lastocc_fill_k<<<(BB * LL + 255) / 256, 256, 0, stream>>>(seq, lastocc);
  masked_softmax_k<<<BB * TT, 256, 0, stream>>>(out, lastocc);
}
